// SparseResBlock_25598005084689
// MI455X (gfx1250) — matrix operand order repaired and checked by reading, not compiled, not run on hardware
//
#include <hip/hip_runtime.h>

typedef __attribute__((ext_vector_type(16))) __bf16 v16bf;
typedef __attribute__((ext_vector_type(8)))  __bf16 v8bf;
typedef __attribute__((ext_vector_type(8)))  float  v8f;

#define N_VOX 131072
#define CCH   128
#define KTAPS 27
#define WELEM (KTAPS * CCH * CCH)   // 442368 elements per weight tensor

// ---------------------------------------------------------------------------
// f32 -> bf16 conversion (activations)
// ---------------------------------------------------------------------------
__global__ void k_cvt_bf16(const float* __restrict__ x, __bf16* __restrict__ y, int n) {
    for (int i = blockIdx.x * blockDim.x + threadIdx.x; i < n; i += gridDim.x * blockDim.x)
        y[i] = (__bf16)x[i];
}

// ---------------------------------------------------------------------------
// Pack W[27][Cin][Cout] f32 into bf16 B-fragment order:
//   dst[(((k*8+nt)*4+kc)*2+half)*16+col)*16 + j] = W[k][kc*32+(j&8)*2+half*8+(j&7)][nt*16+col]
// so each lane's B fragment (16 bf16, K ascending) is one contiguous 32B read.
// ---------------------------------------------------------------------------
__global__ void k_pack_w(const float* __restrict__ w, __bf16* __restrict__ wp) {
    int o = blockIdx.x * 256 + threadIdx.x;       // grid sized to exactly WELEM
    int e  = o & 15;
    int c  = (o >> 4) & 15;
    int h  = (o >> 8) & 1;
    int kc = (o >> 9) & 3;
    int nt = (o >> 11) & 7;
    int k  = o >> 14;
    int cin  = kc * 32 + (e & 8) * 2 + h * 8 + (e & 7);
    int cout = nt * 16 + c;
    wp[o] = (__bf16)w[((size_t)k * CCH + cin) * CCH + cout];
}

// ---------------------------------------------------------------------------
// CDNA5 async global->LDS copy (ASYNCcnt-tracked, no VGPR staging)
// ---------------------------------------------------------------------------
__device__ __forceinline__ void async_lds_b128(unsigned int lds_off, const void* gaddr) {
    asm volatile("global_load_async_to_lds_b128 %0, %1, off"
                 :: "v"(lds_off), "v"(gaddr) : "memory");
}
__device__ __forceinline__ void wait_async0() {
    asm volatile("s_wait_asynccnt 0x0" ::: "memory");
}

// ---------------------------------------------------------------------------
// Gathered submanifold conv as WMMA GEMM.
// Block = 128 threads = 4 waves; block tile = 128 voxels x 128 out-channels.
// Wave tile = 32 voxels x 128 out-channels -> 16 f32 16x16 accumulators;
// each B fragment read from LDS feeds two WMMAs. Weight blocks (32 KB per
// k-tap) are double-buffered in LDS and streamed with async-to-LDS loads
// overlapped with the 64 WMMAs of the current tap.
// ---------------------------------------------------------------------------
__global__ void __launch_bounds__(128)
k_subm_conv_wmma(const __bf16* __restrict__ xin, const int* __restrict__ nbr,
                 const __bf16* __restrict__ wp, float* __restrict__ yout) {
    __shared__ __bf16 lds[2][CCH * CCH];         // 2 x 32 KB double buffer

    const int tid  = threadIdx.x;
    const int wave = tid >> 5;
    const int lane = tid & 31;
    const int half = (lane >> 4) & 1;
    const int col  = lane & 15;                  // A row within tile / D column
    const int vox0 = blockIdx.x * 128 + wave * 32 + col;
    const int vox1 = vox0 + 16;

    // LDS byte offsets of the two buffers (flat LDS aperture: offset = addr[31:0])
    const unsigned int ldsBase0 = (unsigned int)(uintptr_t)(&lds[0][0]);
    const unsigned int ldsBase1 = (unsigned int)(uintptr_t)(&lds[1][0]);

    v8bf z8;
#pragma unroll
    for (int i = 0; i < 8; ++i) z8[i] = (__bf16)0.0f;

    v8f acc0[8], acc1[8];
#pragma unroll
    for (int nt = 0; nt < 8; ++nt)
#pragma unroll
        for (int i = 0; i < 8; ++i) { acc0[nt][i] = 0.0f; acc1[nt][i] = 0.0f; }

    // stage W_k into buffer `buf`: 32 KB / 128 threads = 16 x 16B per thread
#define STAGE_WK(kk, bufbase)                                                   \
    {                                                                           \
        const char* src = (const char*)(wp + (size_t)(kk) * (CCH * CCH)) + tid * 16; \
        unsigned int dst = (bufbase) + tid * 16;                                \
        _Pragma("unroll")                                                       \
        for (int i = 0; i < 16; ++i)                                            \
            async_lds_b128(dst + i * 2048, src + (size_t)i * 2048);             \
    }

    STAGE_WK(0, ldsBase0);
    wait_async0();
    __syncthreads();

    for (int k = 0; k < KTAPS; ++k) {
        const int buf = k & 1;
        if (k + 1 < KTAPS) {                      // prefetch next tap's weights
            STAGE_WK(k + 1, (buf ? ldsBase0 : ldsBase1));
        }

        const int idx0  = nbr[(size_t)vox0 * KTAPS + k];
        const int idx1  = nbr[(size_t)vox1 * KTAPS + k];
        const bool m0   = idx0 >= 0;
        const bool m1   = idx1 >= 0;
        const v8bf* arow0 = (const v8bf*)(xin + (size_t)(m0 ? idx0 : 0) * CCH);
        const v8bf* arow1 = (const v8bf*)(xin + (size_t)(m1 ? idx1 : 0) * CCH);
        const v8bf* bl    = (const v8bf*)&lds[buf][0];

#pragma unroll
        for (int kc = 0; kc < 4; ++kc) {
            // A fragment, ISA layout: lanes 0-15 hold K = kc*32 + {0..7, 16..23},
            //                          lanes 16-31 hold K = kc*32 + {8..15, 24..31}
            v8bf alo0 = m0 ? arow0[kc * 4 + half]     : z8;
            v8bf ahi0 = m0 ? arow0[kc * 4 + half + 2] : z8;
            v8bf alo1 = m1 ? arow1[kc * 4 + half]     : z8;
            v8bf ahi1 = m1 ? arow1[kc * 4 + half + 2] : z8;
            v16bf a0 = __builtin_shufflevector(alo0, ahi0,
                0, 1, 2, 3, 4, 5, 6, 7, 8, 9, 10, 11, 12, 13, 14, 15);
            v16bf a1 = __builtin_shufflevector(alo1, ahi1,
                0, 1, 2, 3, 4, 5, 6, 7, 8, 9, 10, 11, 12, 13, 14, 15);

#pragma unroll
            for (int nt = 0; nt < 8; ++nt) {
                const int bofs8 = ((((nt * 4 + kc) * 2 + half) * 16) + col) * 2; // v8bf units
                v8bf blo = bl[bofs8];
                v8bf bhi = bl[bofs8 + 1];
                v16bf bfrag = __builtin_shufflevector(blo, bhi,
                    0, 1, 2, 3, 4, 5, 6, 7, 8, 9, 10, 11, 12, 13, 14, 15);
                acc0[nt] = __builtin_amdgcn_wmma_f32_16x16x32_bf16(
                    false, a0, false, bfrag, (short)0, acc0[nt], false, false);
                acc1[nt] = __builtin_amdgcn_wmma_f32_16x16x32_bf16(
                    false, a1, false, bfrag, (short)0, acc1[nt], false, false);
            }
        }

        wait_async0();                            // k+1 weights resident in LDS
        __syncthreads();
    }
#undef STAGE_WK

    // D layout: VGPR r -> row M = r + 8*half, column N = nt*16 + (lane&15)
    const int rbase0 = blockIdx.x * 128 + wave * 32 + 8 * half;
    const int rbase1 = rbase0 + 16;
#pragma unroll
    for (int nt = 0; nt < 8; ++nt)
#pragma unroll
        for (int r = 0; r < 8; ++r) {
            yout[(size_t)(rbase0 + r) * CCH + nt * 16 + col] = acc0[nt][r];
            yout[(size_t)(rbase1 + r) * CCH + nt * 16 + col] = acc1[nt][r];
        }
}

// ---------------------------------------------------------------------------
// BN batch statistics: per-column sum and sum-of-squares into stats[256]
// ---------------------------------------------------------------------------
__global__ void k_zero256(float* p) { p[threadIdx.x] = 0.0f; }

__global__ void k_stats(const float* __restrict__ y, float* __restrict__ st) {
    const int c = threadIdx.x;
    float s = 0.0f, q = 0.0f;
    for (int row = blockIdx.x; row < N_VOX; row += gridDim.x) {
        float v = y[(size_t)row * CCH + c];
        s += v; q += v * v;
    }
    atomicAdd(&st[c], s);
    atomicAdd(&st[CCH + c], q);
}

// ---------------------------------------------------------------------------
// Fused BN + LeakyReLU; emits bf16 activations (next conv input) and
// optionally the f32 residual copy.
// ---------------------------------------------------------------------------
__global__ void k_bn_act(const float* __restrict__ y, const float* __restrict__ st,
                         const float* __restrict__ g, const float* __restrict__ b,
                         float slope, float* __restrict__ resOut, __bf16* __restrict__ xbf) {
    const int c = threadIdx.x;
    const float invN = 1.0f / (float)N_VOX;
    const float mu  = st[c] * invN;
    const float var = st[CCH + c] * invN - mu * mu;
    const float sc  = rsqrtf(var + 1e-4f) * g[c];
    const float bi  = b[c];
    for (int row = blockIdx.x; row < N_VOX; row += gridDim.x) {
        size_t o = (size_t)row * CCH + c;
        float v = (y[o] - mu) * sc + bi;
        v = v >= 0.0f ? v : slope * v;
        if (resOut) resOut[o] = v;
        xbf[o] = (__bf16)v;
    }
}

// ---------------------------------------------------------------------------
// Final: BN + residual add + LeakyReLU(1/3), in-place on d_out
// ---------------------------------------------------------------------------
__global__ void k_bn_res_act(const float* __restrict__ y, const float* __restrict__ st,
                             const float* __restrict__ g, const float* __restrict__ b,
                             const float* __restrict__ res, float* out) {
    const int c = threadIdx.x;
    const float invN = 1.0f / (float)N_VOX;
    const float mu  = st[c] * invN;
    const float var = st[CCH + c] * invN - mu * mu;
    const float sc  = rsqrtf(var + 1e-4f) * g[c];
    const float bi  = b[c];
    for (int row = blockIdx.x; row < N_VOX; row += gridDim.x) {
        size_t o = (size_t)row * CCH + c;
        float v = (y[o] - mu) * sc + bi + res[o];
        out[o] = v >= 0.0f ? v : v * (1.0f / 3.0f);
    }
}

// ---------------------------------------------------------------------------
extern "C" void kernel_launch(void* const* d_in, const int* in_sizes, int n_in,
                              void* d_out, int out_size, void* d_ws, size_t ws_size,
                              hipStream_t stream) {
    const float* features = (const float*)d_in[0];
    const int*   nbr      = (const int*)d_in[1];
    const float* w1 = (const float*)d_in[2];
    const float* w2 = (const float*)d_in[3];
    const float* w3 = (const float*)d_in[4];
    const float* g1 = (const float*)d_in[5];
    const float* b1 = (const float*)d_in[6];
    const float* g2 = (const float*)d_in[7];
    const float* b2 = (const float*)d_in[8];
    const float* g3 = (const float*)d_in[9];
    const float* b3 = (const float*)d_in[10];
    float* y = (float*)d_out;                       // also used as f32 conv scratch

    // workspace layout (256B-aligned regions)
    char* base = (char*)d_ws;
    const size_t BF_BYTES = (size_t)N_VOX * CCH * 2;   // 32 MB
    const size_t F_BYTES  = (size_t)N_VOX * CCH * 4;   // 64 MB
    const size_t WP_BYTES = (size_t)WELEM * 2;         // ~864 KB
    __bf16* xa  = (__bf16*)base;
    __bf16* xb  = (__bf16*)(base + BF_BYTES);
    float*  res = (float*)(base + 2 * BF_BYTES);
    __bf16* wp1 = (__bf16*)(base + 2 * BF_BYTES + F_BYTES);
    __bf16* wp2 = (__bf16*)(base + 2 * BF_BYTES + F_BYTES + WP_BYTES);
    __bf16* wp3 = (__bf16*)(base + 2 * BF_BYTES + F_BYTES + 2 * WP_BYTES);
    float*  st  = (float*)(base + 2 * BF_BYTES + F_BYTES + 3 * WP_BYTES);

    // conversions / weight packing
    k_cvt_bf16<<<4096, 256, 0, stream>>>(features, xa, N_VOX * CCH);
    k_pack_w<<<WELEM / 256, 256, 0, stream>>>(w1, wp1);
    k_pack_w<<<WELEM / 256, 256, 0, stream>>>(w2, wp2);
    k_pack_w<<<WELEM / 256, 256, 0, stream>>>(w3, wp3);

    // conv1 -> BN -> LReLU(0.05); save residual (f32) + bf16 input for conv2
    k_zero256<<<1, 256, 0, stream>>>(st);
    k_subm_conv_wmma<<<N_VOX / 128, 128, 0, stream>>>(xa, nbr, wp1, y);
    k_stats<<<1024, 128, 0, stream>>>(y, st);
    k_bn_act<<<1024, 128, 0, stream>>>(y, st, g1, b1, 0.05f, res, xb);

    // conv2 -> BN -> LReLU(0.05); bf16 input for conv3
    k_zero256<<<1, 256, 0, stream>>>(st);
    k_subm_conv_wmma<<<N_VOX / 128, 128, 0, stream>>>(xb, nbr, wp2, y);
    k_stats<<<1024, 128, 0, stream>>>(y, st);
    k_bn_act<<<1024, 128, 0, stream>>>(y, st, g2, b2, 0.05f, nullptr, xa);

    // conv3 -> BN -> +res -> LReLU(1/3), in-place on d_out
    k_zero256<<<1, 256, 0, stream>>>(st);
    k_subm_conv_wmma<<<N_VOX / 128, 128, 0, stream>>>(xa, nbr, wp3, y);
    k_stats<<<1024, 128, 0, stream>>>(y, st);
    k_bn_res_act<<<1024, 128, 0, stream>>>(y, st, g3, b3, res, y);
}
